// DetectionDecoder_44246753084174
// MI455X (gfx1250) — matrix operand order repaired and checked by reading, not compiled, not run on hardware
//
#include <hip/hip_runtime.h>
#include <hip/hip_bf16.h>

typedef __bf16 bf16;
typedef bf16  bf16x8  __attribute__((ext_vector_type(8)));
typedef bf16  bf16x16 __attribute__((ext_vector_type(16)));
typedef float f32x8   __attribute__((ext_vector_type(8)));

#define HIDDEN 1024
#define IN_DIM 12544   // 256 * 49
#define KBOX   1024    // B*N
#define NCLS   80
#define MN     (1024 * 1024)   // GEMM output elements (M=N=1024)

// ---------------- workspace layout (bytes) ----------------
static constexpr size_t WS_A    = 0;                                  // bf16 [1024][12544] activations
static constexpr size_t WS_W1T  = WS_A    + (size_t)KBOX  *IN_DIM*2;  // bf16 [1024][12544] w1^T
static constexpr size_t WS_W2T  = WS_W1T  + (size_t)HIDDEN*IN_DIM*2;  // bf16 [1024][1024]  w2^T
static constexpr size_t WS_H1   = WS_W2T  + (size_t)HIDDEN*HIDDEN*2;  // bf16 [1024][1024]
static constexpr size_t WS_H2   = WS_H1   + (size_t)KBOX  *HIDDEN*2;  // f32  [1024][1024]
static constexpr size_t WS_TEMB = WS_H2   + (size_t)KBOX  *HIDDEN*4;  // f32  [4][1024]
static constexpr size_t WS_PART = WS_TEMB + (size_t)4*HIDDEN*4;       // f32  [4][1024][1024] split-K partials

// ---------------- f32 -> bf16 transposed weight convert ----------------
__global__ __launch_bounds__(256)
void transpose_bf16_kernel(const float* __restrict__ W, bf16* __restrict__ Wt, int K, int N) {
    __shared__ float tile[32][33];
    const int n0 = blockIdx.x * 32, k0 = blockIdx.y * 32;
    const int tx = threadIdx.x & 31, ty = threadIdx.x >> 5;   // 32 x 8
    #pragma unroll
    for (int r = ty; r < 32; r += 8)
        tile[r][tx] = W[(size_t)(k0 + r) * N + n0 + tx];
    __syncthreads();
    #pragma unroll
    for (int r = ty; r < 32; r += 8)
        Wt[(size_t)(n0 + r) * K + k0 + tx] = (bf16)tile[tx][r];
}

// ---------------- time embedding MLP: temb[b][n] = relu(emb(t_b) @ wt + bt) ----------------
__global__ __launch_bounds__(256)
void temb_kernel(const float* __restrict__ t, const float* __restrict__ wt,
                 const float* __restrict__ bt, float* __restrict__ temb) {
    const int n = blockIdx.x * 256 + threadIdx.x;   // 0..1023
    const int b = blockIdx.y;                       // 0..3
    const float tv = t[b];
    float s = bt[n];
    const float cexp = -9.210340371976184f / 64.0f; // -ln(10000)/half
    for (int i = 0; i < 64; ++i) {
        float ang = tv * __expf(cexp * (float)i);
        s += __sinf(ang) * wt[i * HIDDEN + n] + __cosf(ang) * wt[(64 + i) * HIDDEN + n];
    }
    temb[b * HIDDEN + n] = fmaxf(s, 0.0f);
}

// ---------------- ROI align ----------------
__device__ __forceinline__ void axis_interp(float co, int size, int& lo, int& hi,
                                            float& fr, bool& valid) {
    valid = (co >= -1.0f) && (co <= (float)size);
    float c = fmaxf(co, 0.0f);
    float lowf = floorf(c);
    lo = (int)lowf; if (lo > size - 1) lo = size - 1;
    hi = lo + 1;    if (hi > size - 1) hi = size - 1;
    if (lowf >= (float)(size - 1)) c = (float)(size - 1);
    fr = c - (float)lo;
}

__global__ __launch_bounds__(256)
void roi_align_kernel(const float* __restrict__ boxes_t,
                      const float* __restrict__ f0, const float* __restrict__ f1,
                      const float* __restrict__ f2, const float* __restrict__ f3,
                      bf16* __restrict__ A) {
    const int m = blockIdx.x;           // box index 0..1023
    const int b = m >> 8;               // batch
    const float cx = boxes_t[m * 4 + 0], cy = boxes_t[m * 4 + 1];
    const float w  = boxes_t[m * 4 + 2], h  = boxes_t[m * 4 + 3];
    const float x1 = cx - 0.5f * w, y1 = cy - 0.5f * h;
    const float x2 = cx + 0.5f * w, y2 = cy + 0.5f * h;
    const float area = (x2 - x1) * (y2 - y1);
    float lvl = floorf(4.0f + log2f(sqrtf(area) / 224.0f + 1e-6f));
    lvl = fminf(fmaxf(lvl, 2.0f), 5.0f);
    const int idx = (int)lvl - 2;                 // 0..3
    const float scale = 0.25f / (float)(1 << idx);
    const int HW = 256 >> idx;
    const float* feat = (idx == 0) ? f0 : (idx == 1) ? f1 : (idx == 2) ? f2 : f3;
    const float sx1 = x1 * scale, sy1 = y1 * scale;
    const float bw = fmaxf((x2 - x1) * scale, 1.0f) * (1.0f / 7.0f);
    const float bh = fmaxf((y2 - y1) * scale, 1.0f) * (1.0f / 7.0f);
    const size_t fbase = (size_t)b * 256 * HW * HW;

    for (int it = 0; it < 49; ++it) {
        const int k = it * 256 + (int)threadIdx.x;    // 0..12543, contiguous -> coalesced store
        const int c = k / 49;
        const int cell = k - c * 49;
        const int gy = cell / 7, gx = cell - gy * 7;
        const float* fc = feat + fbase + (size_t)c * HW * HW;
        float acc = 0.0f;
        #pragma unroll
        for (int sy = 0; sy < 2; ++sy) {
            const float yc = sy1 + ((float)(2 * gy + sy) + 0.5f) * 0.5f * bh;
            int ylo, yhi; float fy; bool vy;
            axis_interp(yc, HW, ylo, yhi, fy, vy);
            #pragma unroll
            for (int sx = 0; sx < 2; ++sx) {
                const float xc = sx1 + ((float)(2 * gx + sx) + 0.5f) * 0.5f * bw;
                int xlo, xhi; float fx; bool vx;
                axis_interp(xc, HW, xlo, xhi, fx, vx);
                if (vy && vx) {
                    const float v00 = fc[ylo * HW + xlo], v01 = fc[ylo * HW + xhi];
                    const float v10 = fc[yhi * HW + xlo], v11 = fc[yhi * HW + xhi];
                    const float hy = 1.0f - fy, hx = 1.0f - fx;
                    acc += hy * hx * v00 + hy * fx * v01 + fy * hx * v10 + fy * fx * v11;
                }
            }
        }
        A[(size_t)m * IN_DIM + k] = (bf16)(acc * 0.25f);
    }
}

// ---------------- split-K WMMA GEMM: Cpart[z] = A[:, zK:zK+klen] @ Bt[:, zK:zK+klen]^T ----------------
// M = N = 1024 fixed. 128x128 block tile, 256 threads = 8 waves (2 x 4),
// wave tile 64x32 (4x2 WMMA tiles). Double-buffered LDS + register prefetch.
__global__ __launch_bounds__(256)
void gemm_wmma_part_kernel(const bf16* __restrict__ A, const bf16* __restrict__ Bt,
                           float* __restrict__ Cpart, int K, int klen) {
    __shared__ bf16 lA[2][128 * 32];   // A tile, row-major [m][k]
    __shared__ bf16 lB[2][128 * 32];   // B tile, N-major  [n][k]
    const int tid  = threadIdx.x;
    const int lane = tid & 31, wave = tid >> 5;
    const int waveM = wave >> 2, waveN = wave & 3;  // 2 x 4 wave grid
    const int l16 = lane & 15, lh = lane >> 4;
    const int bm = blockIdx.y * 128, bn = blockIdx.x * 128;
    const int kbeg = blockIdx.z * klen, kend = kbeg + klen;

    // staging: 512 x 16B chunks per tile, 2 chunks per thread per tile
    const int r0 = tid >> 2,        c0 = tid & 3;
    const int r1 = (tid + 256) >> 2, c1 = tid & 3;   // r1 = r0 + 64
    uint4 rA0, rA1, rB0, rB1;

    auto load_regs = [&](int kk) {
        rA0 = *(const uint4*)(A  + (size_t)(bm + r0) * K + kk + c0 * 8);
        rA1 = *(const uint4*)(A  + (size_t)(bm + r1) * K + kk + c1 * 8);
        rB0 = *(const uint4*)(Bt + (size_t)(bn + r0) * K + kk + c0 * 8);
        rB1 = *(const uint4*)(Bt + (size_t)(bn + r1) * K + kk + c1 * 8);
    };
    auto store_lds = [&](int buf) {
        *(uint4*)&lA[buf][r0 * 32 + c0 * 8] = rA0;
        *(uint4*)&lA[buf][r1 * 32 + c1 * 8] = rA1;
        *(uint4*)&lB[buf][r0 * 32 + c0 * 8] = rB0;
        *(uint4*)&lB[buf][r1 * 32 + c1 * 8] = rB1;
    };

    f32x8 acc[4][2] = {};

    load_regs(kbeg);
    store_lds(0);
    if (kbeg + 32 < kend) load_regs(kbeg + 32);    // prefetch tile 1
    __syncthreads();

    int cur = 0;
    for (int kk = kbeg; kk < kend; kk += 32) {
        // compute from LDS buffer `cur`
        bf16x16 af[4], bfg[2];
        #pragma unroll
        for (int i = 0; i < 4; ++i) {
            // A frag: lane = M; lanes 0-15 hold K 0-7 & 16-23, lanes 16-31 hold K 8-15 & 24-31
            const bf16* p = &lA[cur][(waveM * 64 + i * 16 + l16) * 32];
            bf16x8 a0 = *(const bf16x8*)(p + lh * 8);
            bf16x8 a1 = *(const bf16x8*)(p + 16 + lh * 8);
            af[i] = __builtin_shufflevector(a0, a1, 0,1,2,3,4,5,6,7,8,9,10,11,12,13,14,15);
        }
        #pragma unroll
        for (int j = 0; j < 2; ++j) {
            // B frag: lane = N; lanes 0-15 hold K 0-7 & 16-23, lanes 16-31 hold K 8-15 & 24-31
            const bf16* p = &lB[cur][(waveN * 32 + j * 16 + l16) * 32];
            bf16x8 b0 = *(const bf16x8*)(p + lh * 8);
            bf16x8 b1 = *(const bf16x8*)(p + 16 + lh * 8);
            bfg[j] = __builtin_shufflevector(b0, b1, 0,1,2,3,4,5,6,7,8,9,10,11,12,13,14,15);
        }
        #pragma unroll
        for (int i = 0; i < 4; ++i)
            #pragma unroll
            for (int j = 0; j < 2; ++j)
                acc[i][j] = __builtin_amdgcn_wmma_f32_16x16x32_bf16(
                    false, af[i], false, bfg[j], (short)0, acc[i][j], false, false);

        // stage next tile into the idle buffer while this one is consumed
        if (kk + 32 < kend) store_lds(1 - cur);
        __syncthreads();
        if (kk + 64 < kend) load_regs(kk + 64);    // prefetch tile after next
        cur ^= 1;
    }

    // write raw f32 partial: C layout — VGPR r: M = lh*8 + r, N = l16
    float* Cp = Cpart + (size_t)blockIdx.z * MN;
    #pragma unroll
    for (int i = 0; i < 4; ++i) {
        #pragma unroll
        for (int j = 0; j < 2; ++j) {
            const int n = bn + waveN * 32 + j * 16 + l16;
            #pragma unroll
            for (int r = 0; r < 8; ++r) {
                const int m = bm + waveM * 64 + i * 16 + lh * 8 + r;
                Cp[(size_t)m * 1024 + n] = acc[i][j][r];
            }
        }
    }
}

// ---------------- split-K reduce + fused epilogues ----------------
// GEMM1: h1 = relu(sum_z part + b1) + temb[m>>8]   (bf16)
__global__ __launch_bounds__(256)
void epilogue1_kernel(const float* __restrict__ part, const float* __restrict__ bias,
                      const float* __restrict__ temb, bf16* __restrict__ h1) {
    const int idx = blockIdx.x * 256 + threadIdx.x;   // 0 .. MN-1
    const int m = idx >> 10, n = idx & 1023;
    float s = part[idx] + part[idx + MN] + part[idx + 2 * MN] + part[idx + 3 * MN];
    s = fmaxf(s + bias[n], 0.0f) + temb[(m >> 8) * HIDDEN + n];
    h1[idx] = (bf16)s;
}

// GEMM2: h2 = relu(sum_z part + b2)   (f32)
__global__ __launch_bounds__(256)
void epilogue2_kernel(const float* __restrict__ part, const float* __restrict__ bias,
                      float* __restrict__ h2) {
    const int idx = blockIdx.x * 256 + threadIdx.x;
    const int n = idx & 1023;
    float s = part[idx] + part[idx + MN];
    h2[idx] = fmaxf(s + bias[n], 0.0f);
}

// ---------------- heads: delta (4) + logits (80) per box ----------------
__global__ __launch_bounds__(128)
void head_kernel(const float* __restrict__ h2,
                 const float* __restrict__ wreg, const float* __restrict__ breg,
                 const float* __restrict__ wcls, const float* __restrict__ bcls,
                 const float* __restrict__ boxes_t, float* __restrict__ out) {
    __shared__ float row[HIDDEN];
    const int m = blockIdx.x;
    for (int k = threadIdx.x; k < HIDDEN; k += 128) row[k] = h2[(size_t)m * HIDDEN + k];
    __syncthreads();
    const int t = threadIdx.x;
    if (t < 4) {
        float s = breg[t];
        for (int k = 0; k < HIDDEN; ++k) s += row[k] * wreg[k * 4 + t];
        out[m * 4 + t] = boxes_t[m * 4 + t] + s;
    } else if (t < 4 + NCLS) {
        const int j = t - 4;
        float s = bcls[j];
        for (int k = 0; k < HIDDEN; ++k) s += row[k] * wcls[k * NCLS + j];
        out[4096 + (size_t)m * NCLS + j] = s;
    }
}

extern "C" void kernel_launch(void* const* d_in, const int* in_sizes, int n_in,
                              void* d_out, int out_size, void* d_ws, size_t ws_size,
                              hipStream_t stream) {
    const float* boxes_t = (const float*)d_in[0];
    const float* t       = (const float*)d_in[1];
    const float* f0      = (const float*)d_in[2];
    const float* f1      = (const float*)d_in[3];
    const float* f2      = (const float*)d_in[4];
    const float* f3      = (const float*)d_in[5];
    const float* w1      = (const float*)d_in[6];
    const float* b1      = (const float*)d_in[7];
    const float* wt      = (const float*)d_in[8];
    const float* bt      = (const float*)d_in[9];
    const float* w2      = (const float*)d_in[10];
    const float* b2      = (const float*)d_in[11];
    const float* wreg    = (const float*)d_in[12];
    const float* breg    = (const float*)d_in[13];
    const float* wcls    = (const float*)d_in[14];
    const float* bcls    = (const float*)d_in[15];

    char* ws = (char*)d_ws;
    bf16*  A    = (bf16*)(ws + WS_A);
    bf16*  W1t  = (bf16*)(ws + WS_W1T);
    bf16*  W2t  = (bf16*)(ws + WS_W2T);
    bf16*  h1   = (bf16*)(ws + WS_H1);
    float* h2   = (float*)(ws + WS_H2);
    float* temb = (float*)(ws + WS_TEMB);
    float* part = (float*)(ws + WS_PART);

    // independent producers (stream-ordered)
    transpose_bf16_kernel<<<dim3(HIDDEN / 32, IN_DIM / 32), 256, 0, stream>>>(w1, W1t, IN_DIM, HIDDEN);
    transpose_bf16_kernel<<<dim3(HIDDEN / 32, HIDDEN / 32), 256, 0, stream>>>(w2, W2t, HIDDEN, HIDDEN);
    temb_kernel<<<dim3(4, 4), 256, 0, stream>>>(t, wt, bt, temb);
    roi_align_kernel<<<dim3(KBOX), 256, 0, stream>>>(boxes_t, f0, f1, f2, f3, A);

    // GEMM1 (split-K = 4 -> 256 WGs), then fused reduce+bias+relu+temb
    gemm_wmma_part_kernel<<<dim3(8, 8, 4), 256, 0, stream>>>(A, W1t, part, IN_DIM, IN_DIM / 4);
    epilogue1_kernel<<<dim3(MN / 256), 256, 0, stream>>>(part, b1, temb, h1);

    // GEMM2 (split-K = 2 -> 128 WGs), then fused reduce+bias+relu
    gemm_wmma_part_kernel<<<dim3(8, 8, 2), 256, 0, stream>>>(h1, W2t, part, HIDDEN, HIDDEN / 2);
    epilogue2_kernel<<<dim3(MN / 256), 256, 0, stream>>>(part, b2, h2);

    // heads + output assembly
    head_kernel<<<dim3(KBOX), 128, 0, stream>>>(h2, wreg, breg, wcls, bcls, boxes_t, (float*)d_out);
}
